// EEGConvLayer_5944234738392
// MI455X (gfx1250) — matrix operand order repaired and checked by reading, not compiled, not run on hardware
//
#include <hip/hip_runtime.h>
#include <hip/hip_bf16.h>

// B=128, NB=5, N=64, D=512, K=3, GROUP_IN=4
#define Bv 128
#define NBv 5
#define Nv 64
#define Dv 512

typedef __attribute__((ext_vector_type(8)))  _Float16 v8h;
typedef __attribute__((ext_vector_type(16))) _Float16 v16h;
typedef __attribute__((ext_vector_type(8)))  float    v8f;

// ---------------------------------------------------------------------------
// K0: convert proj_w (f32, [d3][d2] row-major, 512x512) to f16
// ---------------------------------------------------------------------------
__global__ void cvt_w_kernel(const float* __restrict__ w, _Float16* __restrict__ wh) {
    int i = blockIdx.x * 256 + threadIdx.x;
    wh[i] = (_Float16)w[i];
}

// ---------------------------------------------------------------------------
// K1: grouped conv along N, writing W1 in the "T" flat layout:
//     W1[((b*D + d)*N + n)*NB + nb] = conv_output[b,nb,n,d]
// block = (b, nb, d-tile of 64); 256 threads
// ---------------------------------------------------------------------------
__global__ __launch_bounds__(256)
void conv_kernel(const float* __restrict__ x, const float* __restrict__ sw,
                 const float* __restrict__ sb, float* __restrict__ W1) {
    int blk = blockIdx.x;
    int dt = blk & 7;
    int nb = (blk >> 3) % NBv;
    int b  = blk / (NBv * 8);
    int d0 = dt * 64;

    __shared__ float s[64 * 65];   // s[n][dl], padded
    const float* xb = x + ((size_t)(b * NBv + nb)) * Nv * Dv;

    for (int t = threadIdx.x; t < 64 * 64; t += 256) {
        int n = t >> 6, dl = t & 63;
        s[n * 65 + dl] = xb[n * Dv + d0 + dl];
    }
    __syncthreads();

    int dl = threadIdx.x & 63;
    int nq = threadIdx.x >> 6;      // 0..3, 16 n-values each
    int d  = d0 + dl;
    int cb4 = dl & ~3;              // group base within tile

    float w[12];
    const float* wp = sw + ((size_t)nb * Dv + d) * 12;   // [nb][d][i][k]
    #pragma unroll
    for (int j = 0; j < 12; ++j) w[j] = wp[j];
    float bias = sb[nb * Dv + d];

    float* outp = W1 + (((size_t)b * Dv + d) * Nv) * NBv + nb;
    for (int nn = 0; nn < 16; ++nn) {
        int n = nq * 16 + nn;
        float acc = bias;
        #pragma unroll
        for (int i = 0; i < 4; ++i) {
            float xm = (n > 0)  ? s[(n - 1) * 65 + cb4 + i] : 0.0f;
            float x0 =            s[n * 65 + cb4 + i];
            float xp = (n < 63) ? s[(n + 1) * 65 + cb4 + i] : 0.0f;
            acc += xm * w[i * 3 + 0] + x0 * w[i * 3 + 1] + xp * w[i * 3 + 2];
        }
        outp[(size_t)n * NBv] = acc;
    }
}

// ---------------------------------------------------------------------------
// K2: cross-band mix in flat space + transpose to GEMM-A layout (f16):
//   F2[g] = sum_m wc[o][m]*W1[(r*NB+m)*D+e] + cb[o],  g=(r*NB+o)*D+e
//   ROWS[q*D + d2] = F2[((b2*D+d2)*NB+m2)*N + n2],  q=(b2*N+n2)*NB+m2
// block = (b2, m2, d2-tile of 64); 256 threads
// ---------------------------------------------------------------------------
__global__ __launch_bounds__(256)
void cross_kernel(const float* __restrict__ W1, const float* __restrict__ cw,
                  const float* __restrict__ cb, _Float16* __restrict__ ROWS) {
    int blk = blockIdx.x;
    int dt = blk & 7;
    int m2 = (blk >> 3) % NBv;
    int b2 = blk / (NBv * 8);
    int d0 = dt * 64;

    __shared__ float s[64 * 65];   // s[dl][n2]
    __shared__ float wc_s[25];
    __shared__ float cb_s[5];
    if (threadIdx.x < 25) wc_s[threadIdx.x] = cw[threadIdx.x * 3 + 1]; // cross_w[o][m][1]
    if (threadIdx.x < 5)  cb_s[threadIdx.x] = cb[threadIdx.x];
    __syncthreads();

    // phase 1: read-coalesced cross mix
    for (int t = threadIdx.x; t < 4096; t += 256) {
        int dl = t >> 6, n2 = t & 63;
        int g = (((b2 * Dv + d0 + dl)) * NBv + m2) * Nv + n2;   // < 2^25
        int e  = g & (Dv - 1);
        int gd = g >> 9;           // g / D
        int o  = gd % NBv;
        int r  = gd / NBv;
        const float* base = W1 + (size_t)r * (NBv * Dv) + e;
        float acc = cb_s[o];
        #pragma unroll
        for (int m = 0; m < NBv; ++m) acc += wc_s[o * NBv + m] * base[(size_t)m * Dv];
        s[dl * 65 + n2] = acc;
    }
    __syncthreads();

    // phase 2: transposed, write-coalesced f16 store
    for (int t = threadIdx.x; t < 4096; t += 256) {
        int n2 = t >> 6, dl = t & 63;
        size_t q = ((size_t)b2 * Nv + n2) * NBv + m2;
        ROWS[q * Dv + d0 + dl] = (_Float16)s[dl * 65 + n2];
    }
}

// ---------------------------------------------------------------------------
// K3: WMMA GEMM  P[q,d3] = ROWS[q,:] . proj_w[d3,:]  (f16 in, f32 acc)
//     fused epilogue: t = P + proj_b + x ; LayerNorm over d3 ; exact GELU.
// block = 64 rows x 512 cols, 8 waves, wave = 64 rows x 64 cols (4x4 WMMA tiles)
// A-frag per lane: row M = lane%16; halves h: K = h + 16*(lane/16)
// B-frag per lane: col N = lane%16; halves h: K = h + 16*(lane/16)
// C/D: lane = N%16 + 16*(M>=8); vgpr v = M%8
// ---------------------------------------------------------------------------
__global__ __launch_bounds__(256)
void gemm_ln_gelu_kernel(const _Float16* __restrict__ A, const _Float16* __restrict__ Bw,
                         const float* __restrict__ x, const float* __restrict__ proj_b,
                         const float* __restrict__ ln_g, const float* __restrict__ ln_b,
                         float* __restrict__ OUT) {
    const int wave  = threadIdx.x >> 5;
    const int lane  = threadIdx.x & 31;
    const int lhalf = lane >> 4;           // 0/1
    const int l16   = lane & 15;
    const int qbase = blockIdx.x * 64;
    const int cbase = wave * 64;

    v8f c[4][4];
    #pragma unroll
    for (int i = 0; i < 4; ++i)
        #pragma unroll
        for (int j = 0; j < 4; ++j)
            c[i][j] = (v8f){0.f, 0.f, 0.f, 0.f, 0.f, 0.f, 0.f, 0.f};

    const int akoff = 16 * lhalf;          // A K sub-offset per lane half
    const int bkoff = 16 * lhalf;          // B K sub-offset per lane half

    for (int kk = 0; kk < Dv; kk += 32) {
        v16h a[4], b[4];
        #pragma unroll
        for (int i = 0; i < 4; ++i) {
            const _Float16* p = A + (size_t)(qbase + i * 16 + l16) * Dv + kk + akoff;
            v8h lo = *(const v8h*)p;
            v8h hi = *(const v8h*)(p + 8);
            a[i] = __builtin_shufflevector(lo, hi, 0,1,2,3,4,5,6,7,8,9,10,11,12,13,14,15);
        }
        #pragma unroll
        for (int j = 0; j < 4; ++j) {
            const _Float16* p = Bw + (size_t)(cbase + j * 16 + l16) * Dv + kk + bkoff;
            b[j] = *(const v16h*)p;        // 32B contiguous: K = kk+bkoff .. +15
        }
        #pragma unroll
        for (int i = 0; i < 4; ++i)
            #pragma unroll
            for (int j = 0; j < 4; ++j)
                c[i][j] = __builtin_amdgcn_wmma_f32_16x16x32_f16(
                    false, a[i], false, b[j], (short)0, c[i][j], false, false);
    }

    // ---- epilogue: residual + proj_b, row statistics ----
    __shared__ float red[2][64][8];        // [sum|sumsq][row][wave]
    __shared__ float mu_s[64], rstd_s[64];

    float pb[4];
    #pragma unroll
    for (int j = 0; j < 4; ++j) pb[j] = proj_b[cbase + j * 16 + l16];

    #pragma unroll
    for (int i = 0; i < 4; ++i) {
        #pragma unroll
        for (int v = 0; v < 8; ++v) {
            int row = i * 16 + v + 8 * lhalf;
            float s1 = 0.f, s2 = 0.f;
            #pragma unroll
            for (int j = 0; j < 4; ++j) {
                float t = c[i][j][v] + pb[j]
                        + x[(size_t)(qbase + row) * Dv + cbase + j * 16 + l16];
                c[i][j][v] = t;
                s1 += t; s2 += t * t;
            }
            // reduce over the 16 lanes (cols) holding this row; masks stay in half
            #pragma unroll
            for (int m = 1; m < 16; m <<= 1) {
                s1 += __shfl_xor(s1, m, 32);
                s2 += __shfl_xor(s2, m, 32);
            }
            if (l16 == 0) { red[0][row][wave] = s1; red[1][row][wave] = s2; }
        }
    }
    __syncthreads();

    if (threadIdx.x < 64) {
        float s1 = 0.f, s2 = 0.f;
        #pragma unroll
        for (int w2 = 0; w2 < 8; ++w2) { s1 += red[0][threadIdx.x][w2]; s2 += red[1][threadIdx.x][w2]; }
        float mu  = s1 * (1.0f / Dv);
        float var = s2 * (1.0f / Dv) - mu * mu;
        mu_s[threadIdx.x]   = mu;
        rstd_s[threadIdx.x] = rsqrtf(var + 1e-5f);
    }
    __syncthreads();

    // ---- normalize + exact GELU + store ----
    #pragma unroll
    for (int i = 0; i < 4; ++i) {
        #pragma unroll
        for (int j = 0; j < 4; ++j) {
            int col = cbase + j * 16 + l16;
            float g = ln_g[col], bb = ln_b[col];
            #pragma unroll
            for (int v = 0; v < 8; ++v) {
                int row = i * 16 + v + 8 * lhalf;
                float t = c[i][j][v];
                float y = (t - mu_s[row]) * rstd_s[row] * g + bb;
                float ge = 0.5f * y * (1.0f + erff(y * 0.70710678118654752f));
                OUT[(size_t)(qbase + row) * Dv + col] = ge;
            }
        }
    }
}

// ---------------------------------------------------------------------------
extern "C" void kernel_launch(void* const* d_in, const int* in_sizes, int n_in,
                              void* d_out, int out_size, void* d_ws, size_t ws_size,
                              hipStream_t stream) {
    const float* x         = (const float*)d_in[0];
    const float* spatial_w = (const float*)d_in[1];
    const float* spatial_b = (const float*)d_in[2];
    const float* cross_w   = (const float*)d_in[3];
    const float* cross_b   = (const float*)d_in[4];
    const float* proj_w    = (const float*)d_in[5];
    const float* proj_b    = (const float*)d_in[6];
    const float* ln_g      = (const float*)d_in[7];
    const float* ln_b      = (const float*)d_in[8];
    float* out = (float*)d_out;

    const size_t total = (size_t)Bv * NBv * Nv * Dv;          // 20,971,520
    char* ws = (char*)d_ws;
    float*    W1   = (float*)ws;                              // 84 MB
    _Float16* ROWS = (_Float16*)(ws + total * 4);             // 42 MB
    _Float16* BwH  = (_Float16*)(ws + total * 4 + total * 2); // 0.5 MB

    cvt_w_kernel<<<(Dv * Dv) / 256, 256, 0, stream>>>(proj_w, BwH);
    conv_kernel<<<Bv * NBv * 8, 256, 0, stream>>>(x, spatial_w, spatial_b, W1);
    cross_kernel<<<Bv * NBv * 8, 256, 0, stream>>>(W1, cross_w, cross_b, ROWS);
    gemm_ln_gelu_kernel<<<(Bv * NBv * Nv) / 64, 256, 0, stream>>>(
        ROWS, BwH, x, proj_b, ln_g, ln_b, out);
}
